// SqrllResid_10900626997646
// MI455X (gfx1250) — compile-verified, matrix operand order checked by reading, not run on hardware
//
#include <hip/hip_runtime.h>
#include <hip/hip_bf16.h>
#include <math.h>

// ---------------------------------------------------------------------------
// Types for CDNA5 WMMA (wave32): v_wmma_f32_16x16x32_bf16
// A: 16x32 bf16 (8 VGPRs), B: 32x16 bf16 (8 VGPRs), C/D: 16x16 f32 (8 VGPRs)
// ---------------------------------------------------------------------------
typedef __attribute__((ext_vector_type(16))) __bf16 v16bf;
typedef __attribute__((ext_vector_type(8)))  float  v8f;
typedef __attribute__((ext_vector_type(4)))  unsigned int u32x4;

__device__ __forceinline__ float sigf(float x) { return 1.0f / (1.0f + __expf(-x)); }

// LDS B-tile geometry: 64 (N) x 32 (K) bf16 slice, row padded to 40 elements
// (80 B) so 16B ds_load_b128 chunks stay aligned and bank stride (20 words)
// cycles through 16 distinct banks across the 16 lanes of a fragment read.
#define LDS_STRIDE 40
#define LDS_TILE   (64 * LDS_STRIDE)

// ---------------------------------------------------------------------------
// Fused WMMA GEMM: out = epilogue(A[MxK] @ W1[NxK]^T, A @ W2[NxK]^T)
//   mode 0: val = acc1 (+ b1[col]) (+ resid)
//   mode 1: val = sigmoid(acc1 + b1[col])
//   mode 2: val = acc1 * sigmoid(acc2 + b2[col])
// Block = 256 threads = 8 waves; block tile 128(M) x 64(N); each wave owns a
// 16(M) x 64(N) strip. B k-slices are shared by all 8 waves -> staged in LDS
// via GLOBAL_LOAD_ASYNC_TO_LDS_B128 (ASYNCcnt) with double buffering.
// Grid = (N/64, Mrows/128).
// ---------------------------------------------------------------------------
__global__ __launch_bounds__(256) void wmma_gemm_bf16_ep(
    const __bf16* __restrict__ A,    // Mrows x K, row-major
    const __bf16* __restrict__ W1,   // N x K, row-major (pre-transposed weight)
    const __bf16* __restrict__ W2,   // N x K or nullptr
    const float*  __restrict__ b1,   // [N] or nullptr
    const float*  __restrict__ b2,   // [N] or nullptr
    const float*  __restrict__ resid,// Mrows x N or nullptr
    float*        __restrict__ outF, // Mrows x N or nullptr
    __bf16*       __restrict__ outB, // Mrows x N or nullptr
    int Mrows, int N, int K, int mode)
{
    __shared__ __bf16 shB[2][2][LDS_TILE];   // [double-buffer][W1/W2][tile]

    const int tid  = threadIdx.x;
    const int wave = tid >> 5;
    const int lane = tid & 31;
    const int half = lane >> 4;      // 0 or 1
    const int r16  = lane & 15;

    const int mBase = blockIdx.y * 128 + wave * 16;
    const int nBase = blockIdx.x * 64;
    const bool gated = (mode == 2);

    // Cooperative async-stage mapping: 256 threads x 16B = one 4KB B slice.
    const int srow = tid >> 2;               // 0..63 (row within N strip)
    const int schk = tid & 3;                // 0..3  (16B chunk within 32 k)
    const __bf16* g1 = W1 + (size_t)(nBase + srow) * K + schk * 8;
    const __bf16* g2 = gated ? (W2 + (size_t)(nBase + srow) * K + schk * 8)
                             : g1;
    const int lelem = srow * LDS_STRIDE + schk * 8;

    // Issue async global->LDS copies of the k0 slice into buffer s.
    // LDS offset operand = low 32 bits of flat shared pointer (aperture form).
    auto stage = [&](int s, int k0) {
        unsigned l1 = (unsigned)(size_t)&shB[s][0][lelem];
        asm volatile("global_load_async_to_lds_b128 %0, %1, off"
                     :: "v"(l1), "v"(g1 + k0) : "memory");
        if (gated) {
            unsigned l2 = (unsigned)(size_t)&shB[s][1][lelem];
            asm volatile("global_load_async_to_lds_b128 %0, %1, off"
                         :: "v"(l2), "v"(g2 + k0) : "memory");
        }
    };

    v8f acc1[4] = {};
    v8f acc2[4] = {};

    const __bf16* Arow = A + (size_t)(mBase + r16) * K + half * 8;

    int s = 0;
    stage(0, 0);
    for (int k0 = 0; k0 < K; k0 += 32) {
        const int nxt = k0 + 32;
        if (nxt < K) {
            stage(s ^ 1, nxt);   // prefetch next slice while current finishes
            if (gated) asm volatile("s_wait_asynccnt 0x2" ::: "memory");
            else       asm volatile("s_wait_asynccnt 0x1" ::: "memory");
        } else {
            asm volatile("s_wait_asynccnt 0x0" ::: "memory");
        }
        __syncthreads();         // current slice visible to all 8 waves

        // A fragment: direct per-wave global_load_b128 pair (rows unique/wave)
        v16bf af;
        ((u32x4*)&af)[0] = *(const u32x4*)(Arow + k0);
        ((u32x4*)&af)[1] = *(const u32x4*)(Arow + k0 + 16);

#pragma unroll
        for (int nt = 0; nt < 4; ++nt) {
            const __bf16* bp = &shB[s][0][(nt * 16 + r16) * LDS_STRIDE + half * 8];
            v16bf bf;
            ((u32x4*)&bf)[0] = *(const u32x4*)bp;
            ((u32x4*)&bf)[1] = *(const u32x4*)(bp + 16);
            acc1[nt] = __builtin_amdgcn_wmma_f32_16x16x32_bf16(
                false, af, false, bf, (short)0, acc1[nt], false, false);
        }
        if (gated) {
#pragma unroll
            for (int nt = 0; nt < 4; ++nt) {
                const __bf16* bp = &shB[s][1][(nt * 16 + r16) * LDS_STRIDE + half * 8];
                v16bf bf;
                ((u32x4*)&bf)[0] = *(const u32x4*)bp;
                ((u32x4*)&bf)[1] = *(const u32x4*)(bp + 16);
                acc2[nt] = __builtin_amdgcn_wmma_f32_16x16x32_bf16(
                    false, af, false, bf, (short)0, acc2[nt], false, false);
            }
        }
        __syncthreads();         // all reads done before buffer is re-staged
        s ^= 1;
    }

    // C/D layout: VGPR v holds M = mBase + 8*half + v, N column = nBase + nt*16 + r16
#pragma unroll
    for (int nt = 0; nt < 4; ++nt) {
        const int col = nBase + nt * 16 + r16;
        const float bias1 = b1 ? b1[col] : 0.0f;
        const float bias2 = b2 ? b2[col] : 0.0f;
#pragma unroll
        for (int v = 0; v < 8; ++v) {
            const int m = mBase + 8 * half + v;
            const size_t idx = (size_t)m * N + col;
            float val;
            if (mode == 1) {
                val = sigf(acc1[nt][v] + bias1);
            } else if (mode == 2) {
                val = acc1[nt][v] * sigf(acc2[nt][v] + bias2);
            } else {
                val = acc1[nt][v] + bias1;
                if (resid) val += resid[idx];
            }
            if (outF) outF[idx] = val;
            if (outB) outB[idx] = (__bf16)val;
        }
    }
}

// ---------------------------------------------------------------------------
// LayerNorm over last dim E, one row per block; bf16 output for WMMA feed.
// ---------------------------------------------------------------------------
__global__ __launch_bounds__(256) void ln_bf16_kernel(
    const float* __restrict__ x, const float* __restrict__ w,
    const float* __restrict__ b, __bf16* __restrict__ y, int E)
{
    __shared__ float s1[256];
    __shared__ float s2[256];
    const int row = blockIdx.x;
    const int tid = threadIdx.x;
    const float* xr = x + (size_t)row * E;

    float sum = 0.0f, sq = 0.0f;
    for (int i = tid; i < E; i += 256) {
        const float v = xr[i];
        sum += v; sq += v * v;
    }
    s1[tid] = sum; s2[tid] = sq;
    __syncthreads();
    for (int s = 128; s > 0; s >>= 1) {
        if (tid < s) { s1[tid] += s1[tid + s]; s2[tid] += s2[tid + s]; }
        __syncthreads();
    }
    const float mu  = s1[0] / (float)E;
    const float var = s2[0] / (float)E - mu * mu;
    const float inv = rsqrtf(var + 1e-5f);
    __bf16* yr = y + (size_t)row * E;
    for (int i = tid; i < E; i += 256) {
        yr[i] = (__bf16)((xr[i] - mu) * inv * w[i] + b[i]);
    }
}

// ---------------------------------------------------------------------------
// Cast fp32 [K x N] -> bf16 [N x K] (weight transpose so WMMA B-fragments are
// contiguous along K per output column).
// ---------------------------------------------------------------------------
__global__ __launch_bounds__(256) void castT_kernel(
    const float* __restrict__ in, __bf16* __restrict__ out, int K, int N)
{
    const size_t idx = (size_t)blockIdx.x * blockDim.x + threadIdx.x;
    if (idx >= (size_t)K * N) return;
    const int k = (int)(idx / N);
    const int n = (int)(idx % N);
    out[(size_t)n * K + k] = (__bf16)in[idx];
}

// ---------------------------------------------------------------------------
// Sequential scan over T fused with output gating:
//   h_t = r_t * h_{t-1} + yi_t ;  g_t = softsign(h_t) * og_t (bf16)
// One thread per (b, m) channel; coalesced across m.
// ---------------------------------------------------------------------------
__global__ __launch_bounds__(256) void scan_gate_kernel(
    const float* __restrict__ yi, const float* __restrict__ r,
    const float* __restrict__ og, const float* __restrict__ mem0,
    __bf16* __restrict__ gB, float* __restrict__ mem_out,
    int B, int T, int Mdim)
{
    const int idx = blockIdx.x * blockDim.x + threadIdx.x;
    if (idx >= B * Mdim) return;
    const int b = idx / Mdim;
    const int m = idx % Mdim;

    float h = mem0[idx];
    size_t i = ((size_t)b * T) * Mdim + m;
    for (int t = 0; t < T; ++t, i += Mdim) {
        h = r[i] * h + yi[i];
        const float g = (h / (1.0f + fabsf(h))) * og[i];
        gB[i] = (__bf16)g;
    }
    mem_out[idx] = h;
}

// ---------------------------------------------------------------------------
// Host-side launcher
// ---------------------------------------------------------------------------
extern "C" void kernel_launch(void* const* d_in, const int* in_sizes, int n_in,
                              void* d_out, int out_size, void* d_ws, size_t ws_size,
                              hipStream_t stream) {
    (void)in_sizes; (void)n_in; (void)out_size; (void)ws_size;

    const int B = 4, T = 2048, E = 1024, M = 2048;
    const int R = B * T;   // 8192 token rows

    const float* x     = (const float*)d_in[0];
    const float* mem0  = (const float*)d_in[1];
    const float* ln1_w = (const float*)d_in[2];
    const float* ln1_b = (const float*)d_in[3];
    const float* wf_w  = (const float*)d_in[4];
    const float* wf_b  = (const float*)d_in[5];
    const float* wi_w  = (const float*)d_in[6];
    const float* wig_w = (const float*)d_in[7];
    const float* wig_b = (const float*)d_in[8];
    const float* wog_w = (const float*)d_in[9];
    const float* wog_b = (const float*)d_in[10];
    const float* wo_w  = (const float*)d_in[11];
    const float* ln2_w = (const float*)d_in[12];
    const float* ln2_b = (const float*)d_in[13];
    const float* fwi_w = (const float*)d_in[14];
    const float* fwg_w = (const float*)d_in[15];
    const float* fwg_b = (const float*)d_in[16];
    const float* fwo_w = (const float*)d_in[17];
    const float* fwo_b = (const float*)d_in[18];

    float* out_x   = (float*)d_out;                       // [R, E]
    float* out_mem = (float*)d_out + (size_t)R * E;       // [B, M]

    // Workspace carve-out (256B aligned for 16B vector loads)
    char* wsb = (char*)d_ws;
    size_t off = 0;
    auto alloc = [&](size_t bytes) -> char* {
        char* p = wsb + off;
        off += (bytes + 255) & ~(size_t)255;
        return p;
    };
    __bf16* yB   = (__bf16*)alloc((size_t)R * E * 2);
    __bf16* wiT  = (__bf16*)alloc((size_t)M * E * 2);
    __bf16* wigT = (__bf16*)alloc((size_t)M * E * 2);
    __bf16* wfT  = (__bf16*)alloc((size_t)M * E * 2);
    __bf16* wogT = (__bf16*)alloc((size_t)M * E * 2);
    __bf16* woT  = (__bf16*)alloc((size_t)E * M * 2);
    __bf16* fwiT = (__bf16*)alloc((size_t)M * E * 2);
    __bf16* fwgT = (__bf16*)alloc((size_t)M * E * 2);
    __bf16* fwoT = (__bf16*)alloc((size_t)E * M * 2);
    float*  yi   = (float*) alloc((size_t)R * M * 4);
    float*  rb   = (float*) alloc((size_t)R * M * 4);
    float*  ogb  = (float*) alloc((size_t)R * M * 4);
    __bf16* gB   = (__bf16*)alloc((size_t)R * M * 2);
    float*  x1   = (float*) alloc((size_t)R * E * 4);
    __bf16* zB   = (__bf16*)alloc((size_t)R * E * 2);
    __bf16* z2B  = (__bf16*)alloc((size_t)R * M * 2);

    // 1) cast + transpose all weights to bf16 [N, K]
    {
        const int nEM = (E * M + 255) / 256;
        castT_kernel<<<nEM, 256, 0, stream>>>(wi_w,  wiT,  E, M);
        castT_kernel<<<nEM, 256, 0, stream>>>(wig_w, wigT, E, M);
        castT_kernel<<<nEM, 256, 0, stream>>>(wf_w,  wfT,  E, M);
        castT_kernel<<<nEM, 256, 0, stream>>>(wog_w, wogT, E, M);
        castT_kernel<<<nEM, 256, 0, stream>>>(fwi_w, fwiT, E, M);
        castT_kernel<<<nEM, 256, 0, stream>>>(fwg_w, fwgT, E, M);
        castT_kernel<<<nEM, 256, 0, stream>>>(wo_w,  woT,  M, E);
        castT_kernel<<<nEM, 256, 0, stream>>>(fwo_w, fwoT, M, E);
    }

    // 2) y = LN1(x)  (bf16)
    ln_bf16_kernel<<<R, 256, 0, stream>>>(x, ln1_w, ln1_b, yB, E);

    const dim3 blk(256);
    const dim3 gridM(M / 64, R / 128);   // N = M = 2048
    const dim3 gridE(E / 64, R / 128);   // N = E = 1024

    // 3) yi = (y@wi) * sigmoid(y@wig + wig_b)
    wmma_gemm_bf16_ep<<<gridM, blk, 0, stream>>>(
        yB, wiT, wigT, nullptr, wig_b, nullptr, yi, nullptr, R, M, E, 2);
    // 4) r = sigmoid(y@wf + wf_b)
    wmma_gemm_bf16_ep<<<gridM, blk, 0, stream>>>(
        yB, wfT, nullptr, wf_b, nullptr, nullptr, rb, nullptr, R, M, E, 1);
    // 5) og = sigmoid(y@wog + wog_b)
    wmma_gemm_bf16_ep<<<gridM, blk, 0, stream>>>(
        yB, wogT, nullptr, wog_b, nullptr, nullptr, ogb, nullptr, R, M, E, 1);

    // 6) scan + gate:  g = softsign(h) * og ; mem_out = h[:, -1]
    scan_gate_kernel<<<(B * M + 255) / 256, 256, 0, stream>>>(
        yi, rb, ogb, mem0, gB, out_mem, B, T, M);

    // 7) x1 = x + g @ wo
    wmma_gemm_bf16_ep<<<gridE, blk, 0, stream>>>(
        gB, woT, nullptr, nullptr, nullptr, x, x1, nullptr, R, E, M, 0);

    // 8) z = LN2(x1)  (bf16)
    ln_bf16_kernel<<<R, 256, 0, stream>>>(x1, ln2_w, ln2_b, zB, E);

    // 9) z2 = (z@fwi) * sigmoid(z@fwg + fwg_b)  (bf16 for next GEMM)
    wmma_gemm_bf16_ep<<<gridM, blk, 0, stream>>>(
        zB, fwiT, fwgT, nullptr, fwg_b, nullptr, nullptr, z2B, R, M, E, 2);

    // 10) out = x1 + z2 @ fwo + fwo_b
    wmma_gemm_bf16_ep<<<gridE, blk, 0, stream>>>(
        z2B, fwoT, nullptr, fwo_b, nullptr, x1, out_x, nullptr, R, E, M, 0);
}
